// LinearLoopLayer_21251498180727
// MI455X (gfx1250) — matrix operand order repaired and checked by reading, not compiled, not run on hardware
//
#include <hip/hip_runtime.h>

// ---------------------------------------------------------------------------
// Linear layer: out[b, j] = sum_i x[b,i] * w[j,i] + bias[j]
// M=2048 (from in_sizes), K=4096, N=4096, fp32 in/out.
//
// Two-pass strategy (CDNA5 / gfx1250):
//   Pass 1: one-shot f32 -> bf16 (hi, lo) split planes into d_ws (96 MB).
//   Pass 2: 128x128x32 tiled GEMM, double-buffered LDS fed by
//           GLOBAL_LOAD_ASYNC_TO_LDS_B128 (ASYNCcnt-tracked DMA copies),
//           24 v_wmma_f32_16x16x32_bf16 per wave per K-step (bf16x3).
// Fallback (ws too small): fused 64x64 kernel that converts in-loop.
// ---------------------------------------------------------------------------

#define K_DIM 4096
#define N_DIM 4096
#define LDSS 40   // padded LDS row stride in bf16 elements (80B, 16B-aligned)

typedef __attribute__((ext_vector_type(16))) __bf16          bf16x16;
typedef __attribute__((ext_vector_type(8)))  float           f32x8;
typedef __attribute__((ext_vector_type(4)))  float           f32x4;
typedef __attribute__((ext_vector_type(8)))  unsigned short  u16x8;

union FragCast {
  u16x8   h[2];   // two 16-byte halves (ds_load_b128 x2)
  bf16x16 v;      // 32-byte WMMA operand
};

__device__ __forceinline__ unsigned short f32_to_bf16_rne(float f) {
  unsigned int u = __float_as_uint(f);
  unsigned int r = u + 0x7FFFu + ((u >> 16) & 1u);
  return (unsigned short)(r >> 16);
}
__device__ __forceinline__ float bf16_bits_to_f32(unsigned short h) {
  return __uint_as_float(((unsigned int)h) << 16);
}

// A fragment (16x32 bf16, MxK): lanes 0-15 -> K 0..7 / 16..23,
// lanes 16-31 -> K 8..15 / 24..31 (per CDNA5 ISA 7.12.2).
__device__ __forceinline__ bf16x16 load_a_frag(const unsigned short* base,
                                               int m, int lh) {
  FragCast f;
  const unsigned short* p = base + m * LDSS + 8 * lh;
  f.h[0] = *(const u16x8*)(p);
  f.h[1] = *(const u16x8*)(p + 16);
  return f.v;
}

// B fragment (32x16 bf16, KxN): lane holds column N=lane&15,
// lanes 0-15 -> K 0..7 / 16..23, lanes 16-31 -> K 8..15 / 24..31 (matches A).
__device__ __forceinline__ bf16x16 load_b_frag(const unsigned short* base,
                                               int n, int lh) {
  FragCast f;
  const unsigned short* p = base + n * LDSS + 8 * lh;
  f.h[0] = *(const u16x8*)(p);
  f.h[1] = *(const u16x8*)(p + 16);
  return f.v;
}

#define WMMA_BF16(A, B, C) \
  __builtin_amdgcn_wmma_f32_16x16x32_bf16(false, (A), false, (B), (short)0, (C), false, false)

// Async memory -> LDS copy, 16B per lane, tracked by ASYNCcnt (CDNA5 TDM path).
__device__ __forceinline__ void async_copy_b128(void* lds_dst, const void* gsrc) {
  // LDS aperture generic pointers carry the wave-relative LDS offset in [31:0].
  unsigned lds_off = (unsigned)(size_t)lds_dst;
  asm volatile("global_load_async_to_lds_b128 %0, %1, off"
               :: "v"(lds_off), "v"(gsrc)
               : "memory");
}
__device__ __forceinline__ void wait_asynccnt0() {
  asm volatile("s_wait_asynccnt 0x0" ::: "memory");
}

// ---------------------------------------------------------------------------
// Pass 1: elementwise f32 -> (hi, lo) bf16 planes.
// ---------------------------------------------------------------------------
__global__ __launch_bounds__(256)
void convert_f32_to_bf16hl(const float* __restrict__ src,
                           unsigned short* __restrict__ hi,
                           unsigned short* __restrict__ lo,
                           int n4) {
  int i = blockIdx.x * 256 + threadIdx.x;
  if (i >= n4) return;
  f32x4 v = *(const f32x4*)(src + (size_t)i * 4);
  unsigned short h[4], l[4];
#pragma unroll
  for (int j = 0; j < 4; ++j) {
    h[j] = f32_to_bf16_rne(v[j]);
    float rem = v[j] - bf16_bits_to_f32(h[j]);   // exact (Sterbenz split)
    l[j] = f32_to_bf16_rne(rem);
  }
  uint2 hp, lp;
  hp.x = (unsigned)h[0] | ((unsigned)h[1] << 16);
  hp.y = (unsigned)h[2] | ((unsigned)h[3] << 16);
  lp.x = (unsigned)l[0] | ((unsigned)l[1] << 16);
  lp.y = (unsigned)l[2] | ((unsigned)l[3] << 16);
  *(uint2*)(hi + (size_t)i * 4) = hp;
  *(uint2*)(lo + (size_t)i * 4) = lp;
}

// ---------------------------------------------------------------------------
// Pass 2: 128x128 tile GEMM from pre-split bf16 planes.
//   256 threads = 8 waves (4 M x 2 N); wave sub-tile 32x64 = 2x4 WMMA tiles.
//   BK=32 per step, double-buffered LDS, async global->LDS producer,
//   one barrier per step.
// ---------------------------------------------------------------------------
#define BM2 128
#define BN2 128
#define BK2 32

__global__ __launch_bounds__(256)
void gemm128_bf16x3_wmma(const unsigned short* __restrict__ XH,
                         const unsigned short* __restrict__ XL,
                         const unsigned short* __restrict__ WH,
                         const unsigned short* __restrict__ WL,
                         const float* __restrict__ Bias,
                         float* __restrict__ Out) {
  // [buffer][plane][row*LDSS]  planes: 0=Ahi 1=Alo 2=Bhi 3=Blo  (80 KB total)
  __shared__ __align__(16) unsigned short lds[2][4][BM2 * LDSS];

  const int tid  = threadIdx.x;
  const int lane = tid & 31;
  const int wv   = tid >> 5;     // wave 0..7
  const int wm   = wv >> 1;      // 0..3 : 32-row strip (2 m-tiles)
  const int wn   = wv & 1;       // 0..1 : 64-col strip (4 n-tiles)
  const int lh   = lane >> 4;
  const int ln   = lane & 15;

  const int bm = blockIdx.y * BM2;
  const int bn = blockIdx.x * BN2;

  // Producer: per plane 512 chunks of 8 bf16 (16B); 2 chunks/thread/plane.
  const unsigned short* gsrc[4];
  gsrc[0] = XH + (size_t)bm * K_DIM;
  gsrc[1] = XL + (size_t)bm * K_DIM;
  gsrc[2] = WH + (size_t)bn * K_DIM;
  gsrc[3] = WL + (size_t)bn * K_DIM;

  const int id0 = tid, id1 = tid + 256;
  const int r0 = id0 >> 2, c0 = (id0 & 3) * 8;   // row, k-offset (elements)
  const int r1 = id1 >> 2, c1 = (id1 & 3) * 8;

  const unsigned short* gptr[4][2];
#pragma unroll
  for (int p = 0; p < 4; ++p) {
    gptr[p][0] = gsrc[p] + (size_t)r0 * K_DIM + c0;
    gptr[p][1] = gsrc[p] + (size_t)r1 * K_DIM + c1;
  }
  const int lo0 = r0 * LDSS + c0;
  const int lo1 = r1 * LDSS + c1;

  // --- prologue: slab 0 -> LDS buf 0 (async DMA) ---
#pragma unroll
  for (int p = 0; p < 4; ++p) {
    async_copy_b128(&lds[0][p][lo0], gptr[p][0]);
    async_copy_b128(&lds[0][p][lo1], gptr[p][1]);
  }
  wait_asynccnt0();
  __syncthreads();

  f32x8 acc[2][4];
#pragma unroll
  for (int tm = 0; tm < 2; ++tm)
#pragma unroll
    for (int tn = 0; tn < 4; ++tn) acc[tm][tn] = (f32x8){};

  int cur = 0;
  for (int k0 = 0; k0 < K_DIM; k0 += BK2) {
    const bool has_next = (k0 + BK2) < K_DIM;
    const int  nxt = cur ^ 1;

    // kick off DMA of the next slab into the other buffer before computing
    if (has_next) {
      const int kn = k0 + BK2;
#pragma unroll
      for (int p = 0; p < 4; ++p) {
        async_copy_b128(&lds[nxt][p][lo0], gptr[p][0] + kn);
        async_copy_b128(&lds[nxt][p][lo1], gptr[p][1] + kn);
      }
      if (k0 + 2 * BK2 < K_DIM) {   // global_prefetch_b8 for slab after next
        __builtin_prefetch(gptr[0][0] + k0 + 2 * BK2, 0, 3);
        __builtin_prefetch(gptr[2][0] + k0 + 2 * BK2, 0, 3);
      }
    }

    // --- compute current slab: 24 WMMAs per wave ---
    bf16x16 ah[2], al[2], bh[4], bl[4];
#pragma unroll
    for (int tm = 0; tm < 2; ++tm) {
      const int m = wm * 32 + tm * 16 + ln;
      ah[tm] = load_a_frag(lds[cur][0], m, lh);
      al[tm] = load_a_frag(lds[cur][1], m, lh);
    }
#pragma unroll
    for (int tn = 0; tn < 4; ++tn) {
      const int n = wn * 64 + tn * 16 + ln;
      bh[tn] = load_b_frag(lds[cur][2], n, lh);
      bl[tn] = load_b_frag(lds[cur][3], n, lh);
    }
#pragma unroll
    for (int tn = 0; tn < 4; ++tn) {
#pragma unroll
      for (int tm = 0; tm < 2; ++tm) {
        acc[tm][tn] = WMMA_BF16(ah[tm], bh[tn], acc[tm][tn]);
        acc[tm][tn] = WMMA_BF16(ah[tm], bl[tn], acc[tm][tn]);
        acc[tm][tn] = WMMA_BF16(al[tm], bh[tn], acc[tm][tn]);
      }
    }

    if (has_next) {
      wait_asynccnt0();   // this wave's DMA into buf nxt has landed in LDS
      __syncthreads();    // all waves' DMA visible; cur reads already consumed
      cur = nxt;
    }
  }

  // --- epilogue: D VGPR v -> (M = v + 8*lh, N = ln) of each 16x16 tile ---
#pragma unroll
  for (int tn = 0; tn < 4; ++tn) {
    const int gn = bn + wn * 64 + tn * 16 + ln;
    const float b = Bias[gn];
#pragma unroll
    for (int tm = 0; tm < 2; ++tm) {
#pragma unroll
      for (int v = 0; v < 8; ++v) {
        const int gm = bm + wm * 32 + tm * 16 + v + 8 * lh;
        Out[(size_t)gm * N_DIM + gn] = acc[tm][tn][v] + b;
      }
    }
  }
}

// ---------------------------------------------------------------------------
// Fallback: fused 64x64 kernel (converts f32->bf16 hi/lo in-loop).
// ---------------------------------------------------------------------------
#define BM 64
#define BN 64
#define BK 32

__device__ __forceinline__ void cvt_store4(unsigned short* __restrict__ hi,
                                           unsigned short* __restrict__ lo,
                                           f32x4 v) {
  unsigned short h[4], l[4];
#pragma unroll
  for (int i = 0; i < 4; ++i) {
    h[i] = f32_to_bf16_rne(v[i]);
    float rem = v[i] - bf16_bits_to_f32(h[i]);
    l[i] = f32_to_bf16_rne(rem);
  }
  uint2 hp, lp;
  hp.x = (unsigned)h[0] | ((unsigned)h[1] << 16);
  hp.y = (unsigned)h[2] | ((unsigned)h[3] << 16);
  lp.x = (unsigned)l[0] | ((unsigned)l[1] << 16);
  lp.y = (unsigned)l[2] | ((unsigned)l[3] << 16);
  *(uint2*)hi = hp;
  *(uint2*)lo = lp;
}

__global__ __launch_bounds__(256)
void linear_bf16x3_wmma(const float* __restrict__ X,
                        const float* __restrict__ W,
                        const float* __restrict__ Bias,
                        float* __restrict__ Out) {
  __shared__ __align__(16) unsigned short As_hi[BM * LDSS];
  __shared__ __align__(16) unsigned short As_lo[BM * LDSS];
  __shared__ __align__(16) unsigned short Bs_hi[BN * LDSS];
  __shared__ __align__(16) unsigned short Bs_lo[BN * LDSS];

  const int tid  = threadIdx.x;
  const int lane = tid & 31;
  const int wv   = tid >> 5;
  const int wm   = wv >> 1;
  const int wn   = wv & 1;
  const int lh   = lane >> 4;
  const int ln   = lane & 15;

  const int bm = blockIdx.y * BM;
  const int bn = blockIdx.x * BN;

  const int i0 = tid, i1 = tid + 256;
  const int r0 = i0 >> 3, c0 = (i0 & 7) * 4;
  const int r1 = i1 >> 3, c1 = (i1 & 7) * 4;

  const float* Xa0 = X + (size_t)(bm + r0) * K_DIM + c0;
  const float* Xa1 = X + (size_t)(bm + r1) * K_DIM + c1;
  const float* Wa0 = W + (size_t)(bn + r0) * K_DIM + c0;
  const float* Wa1 = W + (size_t)(bn + r1) * K_DIM + c1;

  f32x8 acc0 = {}, acc1 = {};

  const int am   = wm * 16 + ln;
  const int bnr0 = wn * 32 + ln;
  const int bnr1 = bnr0 + 16;

  for (int k0 = 0; k0 < K_DIM; k0 += BK) {
    f32x4 xa = *(const f32x4*)(Xa0 + k0);
    f32x4 xb = *(const f32x4*)(Xa1 + k0);
    f32x4 wa = *(const f32x4*)(Wa0 + k0);
    f32x4 wb = *(const f32x4*)(Wa1 + k0);

    __syncthreads();
    cvt_store4(&As_hi[r0 * LDSS + c0], &As_lo[r0 * LDSS + c0], xa);
    cvt_store4(&As_hi[r1 * LDSS + c1], &As_lo[r1 * LDSS + c1], xb);
    cvt_store4(&Bs_hi[r0 * LDSS + c0], &Bs_lo[r0 * LDSS + c0], wa);
    cvt_store4(&Bs_hi[r1 * LDSS + c1], &Bs_lo[r1 * LDSS + c1], wb);
    __syncthreads();

    bf16x16 ah  = load_a_frag(As_hi, am, lh);
    bf16x16 al  = load_a_frag(As_lo, am, lh);
    bf16x16 bh0 = load_b_frag(Bs_hi, bnr0, lh);
    bf16x16 bl0 = load_b_frag(Bs_lo, bnr0, lh);
    bf16x16 bh1 = load_b_frag(Bs_hi, bnr1, lh);
    bf16x16 bl1 = load_b_frag(Bs_lo, bnr1, lh);

    acc0 = WMMA_BF16(ah, bh0, acc0);
    acc0 = WMMA_BF16(ah, bl0, acc0);
    acc0 = WMMA_BF16(al, bh0, acc0);
    acc1 = WMMA_BF16(ah, bh1, acc1);
    acc1 = WMMA_BF16(ah, bl1, acc1);
    acc1 = WMMA_BF16(al, bh1, acc1);
  }

  const int gn0 = bn + wn * 32 + ln;
  const int gn1 = gn0 + 16;
  const float bias0 = Bias[gn0];
  const float bias1 = Bias[gn1];
#pragma unroll
  for (int v = 0; v < 8; ++v) {
    const int gm = bm + wm * 16 + v + 8 * lh;
    Out[(size_t)gm * N_DIM + gn0] = acc0[v] + bias0;
    Out[(size_t)gm * N_DIM + gn1] = acc1[v] + bias1;
  }
}

// ---------------------------------------------------------------------------
extern "C" void kernel_launch(void* const* d_in, const int* in_sizes, int n_in,
                              void* d_out, int out_size, void* d_ws, size_t ws_size,
                              hipStream_t stream) {
  const float* x    = (const float*)d_in[0];
  const float* w    = (const float*)d_in[1];
  const float* bias = (const float*)d_in[2];
  float* out = (float*)d_out;

  const int M = in_sizes[0] / K_DIM;   // 2048
  const size_t MK = (size_t)M * K_DIM;
  const size_t NK = (size_t)N_DIM * K_DIM;
  const size_t need = 2 * (MK + NK) * sizeof(unsigned short);  // 96 MB

  if (ws_size >= need && (M % BM2) == 0) {
    unsigned short* XH = (unsigned short*)d_ws;
    unsigned short* XL = XH + MK;
    unsigned short* WH = XL + MK;
    unsigned short* WL = WH + NK;

    const int x4 = (int)(MK / 4);
    const int w4 = (int)(NK / 4);
    convert_f32_to_bf16hl<<<(x4 + 255) / 256, 256, 0, stream>>>(x, XH, XL, x4);
    convert_f32_to_bf16hl<<<(w4 + 255) / 256, 256, 0, stream>>>(w, WH, WL, w4);

    dim3 grid(N_DIM / BN2, M / BM2);   // 32 x 16
    gemm128_bf16x3_wmma<<<grid, 256, 0, stream>>>(XH, XL, WH, WL, bias, out);
  } else {
    dim3 grid(N_DIM / BN, M / BM);     // 64 x 32
    linear_bf16x3_wmma<<<grid, 256, 0, stream>>>(x, w, bias, out);
  }
}
